// StackedSAModuleMSG_77395310674257
// MI455X (gfx1250) — matrix operand order repaired and checked by reading, not compiled, not run on hardware
//
#include <hip/hip_runtime.h>

// ---------------- problem constants (match reference) ----------------
constexpr int B_   = 4;
constexpr int N_   = 8192;
constexpr int M_   = 4096;
constexpr int NQ   = B_ * M_;     // 16384 queries
constexpr int CIN  = 67;          // 3 + 64
constexpr int KP   = 96;          // padded K (multiple of 32)
constexpr int NS0  = 16;
constexpr int NS1  = 32;
constexpr float R0 = 0.1f;
constexpr float R1 = 0.2f;

typedef __attribute__((ext_vector_type(8)))  _Float16 v8h;
typedef __attribute__((ext_vector_type(16))) _Float16 v16h;
typedef __attribute__((ext_vector_type(8)))  float    v8f;

__device__ __forceinline__ v8f zero8f() {
  v8f z;
#pragma unroll
  for (int i = 0; i < 8; ++i) z[i] = 0.0f;
  return z;
}
__device__ __forceinline__ v8h zero8h() {
  v8h z;
#pragma unroll
  for (int i = 0; i < 8; ++i) z[i] = (_Float16)0.0f;
  return z;
}
__device__ __forceinline__ v16h cat16(v8h lo, v8h hi) {
  return __builtin_shufflevector(lo, hi, 0,1,2,3,4,5,6,7,8,9,10,11,12,13,14,15);
}

// ---------------------------------------------------------------------
// Kernel 1: ball query (both scales in one scan) + new_xyz passthrough.
// One wave32 per query. mmcv stack semantics: first ns indices (in point
// order) with d2 < r^2; pad with first found; empty group -> idx = -1.
// ---------------------------------------------------------------------
__global__ void ball_query_kernel(const float* __restrict__ xyz,
                                  const float* __restrict__ new_xyz,
                                  int* __restrict__ idx0,
                                  int* __restrict__ idx1,
                                  float* __restrict__ out_xyz) {
  const int wave = (int)((blockIdx.x * blockDim.x + threadIdx.x) >> 5);
  const int lane = (int)(threadIdx.x & 31);
  if (wave >= NQ) return;
  const int q = wave;
  const int b = q / M_;

  const float qx = new_xyz[q * 3 + 0];
  const float qy = new_xyz[q * 3 + 1];
  const float qz = new_xyz[q * 3 + 2];
  if (lane < 3) out_xyz[q * 3 + lane] = new_xyz[q * 3 + lane];

  const float* xb = xyz + (size_t)b * N_ * 3;

  int c0 = 0, c1 = 0;
  int f0 = 0, f1 = 0;
  const unsigned below = (lane == 0) ? 0u : ((1u << lane) - 1u);

  for (int base = 0; base < N_; base += 32) {
    const int p = base + lane;
    const float dx = xb[p * 3 + 0] - qx;
    const float dy = xb[p * 3 + 1] - qy;
    const float dz = xb[p * 3 + 2] - qz;
    const float d2 = dx * dx + dy * dy + dz * dz;

    const unsigned m1 = (unsigned)__ballot(d2 < R1 * R1);
    const unsigned m0 = (unsigned)__ballot(d2 < R0 * R0);

    if (m0 && c0 < NS0) {
      if (c0 == 0) f0 = base + __builtin_ctz(m0);
      const int slot = c0 + __popc(m0 & below);
      if (((m0 >> lane) & 1u) && slot < NS0) idx0[q * NS0 + slot] = p;
      c0 += __popc(m0);
    }
    if (m1 && c1 < NS1) {
      if (c1 == 0) f1 = base + __builtin_ctz(m1);
      const int slot = c1 + __popc(m1 & below);
      if (((m1 >> lane) & 1u) && slot < NS1) idx1[q * NS1 + slot] = p;
      c1 += __popc(m1);
    }
    if (c0 >= NS0 && c1 >= NS1) break;
  }

  // padding / empty handling
  if (c0 == 0) {
    if (lane < NS0) idx0[q * NS0 + lane] = -1;
  } else {
    const int slot = c0 + lane;
    if (slot < NS0) idx0[q * NS0 + slot] = f0;
  }
  if (c1 == 0) {
    if (lane < NS1) idx1[q * NS1 + lane] = -1;
  } else {
    const int slot = c1 + lane;
    if (slot < NS1) idx1[q * NS1 + slot] = f1;
  }
}

// ---------------------------------------------------------------------
// Kernel 2: gather + MLP(67->64->64, BN+ReLU) + max-pool, WMMA f16.
// One wave32 per query (grid-strided). Channel order permuted so that
// k=0..63 are features, k=64..66 are rel-xyz, k=67..95 zero pad; the
// weight matrix staged in LDS uses the same permutation.
//
// WMMA fragment layouts used here (16-bit, wave32, contiguous k order):
//   A (16x32): lane<16 row=lane, K=0..15;  lane>=16 row=lane-16,
//              K=16..31  -> two 16B LDS loads at koffA, koffA+8.
//   B (32x16): lane<16 col=lane, K=0..15; lane>=16 col=lane-16, K=16..31
//              -> one contiguous 32B (two v8h) LDS load at koffB.
//   C/D (16x16 f32): lane<16 (M=0..7,N=lane), lane>=16 (M=8..15,N=lane-16).
// ---------------------------------------------------------------------
template <int SCALE>
__global__ void sa_mlp_kernel(const float* __restrict__ xyz,
                              const float* __restrict__ new_xyz,
                              const float* __restrict__ feats,
                              const int*   __restrict__ idxArr,
                              const float* __restrict__ W1,
                              const float* __restrict__ g1,
                              const float* __restrict__ b1,
                              const float* __restrict__ W2,
                              const float* __restrict__ g2,
                              const float* __restrict__ b2,
                              float* __restrict__ out_feat) {
  constexpr int S   = (SCALE == 0) ? NS0 : NS1;   // samples per group
  constexpr int XA  = S * KP;                     // halfs of staged X
  constexpr int X2N = 16 * 64;                    // halfs of layer-2 tile

  extern __shared__ char smem_raw[];
  _Float16* w1lds = (_Float16*)smem_raw;          // 64*96 halfs
  _Float16* w2lds = w1lds + 64 * KP;              // 64*64 halfs
  float* bns1 = (float*)(w2lds + 64 * 64);
  float* bnb1 = bns1 + 64;
  float* bns2 = bnb1 + 64;
  float* bnb2 = bns2 + 64;
  _Float16* waveBase = (_Float16*)(bnb2 + 64);

  const int wid  = (int)(threadIdx.x >> 5);
  const int lane = (int)(threadIdx.x & 31);
  _Float16* Xa = waveBase + (size_t)wid * (XA + X2N);
  _Float16* X2 = Xa + XA;

  // ---- stage weights (f16, permuted K) + BN constants into LDS ----
  const float inv_std = rsqrtf(1.0f + 1e-5f);
  for (int f = (int)threadIdx.x; f < 64 * KP; f += (int)blockDim.x) {
    const int o = f / KP, k = f - o * KP;
    float v = 0.0f;
    if (k < 64)      v = W1[o * CIN + 3 + k];        // feature weights
    else if (k < 67) v = W1[o * CIN + (k - 64)];     // xyz weights
    w1lds[o * KP + k] = (_Float16)v;
  }
  for (int f = (int)threadIdx.x; f < 64 * 64; f += (int)blockDim.x)
    w2lds[f] = (_Float16)W2[f];
  if (threadIdx.x < 64) {
    const int c = (int)threadIdx.x;
    bns1[c] = g1[c] * inv_std;  bnb1[c] = b1[c];
    bns2[c] = g2[c] * inv_std;  bnb2[c] = b2[c];
  }
  __syncthreads();

  const int wavesPerBlock = (int)(blockDim.x >> 5);
  const int gwave  = (int)blockIdx.x * wavesPerBlock + wid;
  const int nwaves = (int)gridDim.x * wavesPerBlock;

  const int col   = lane & 15;
  const int koffA = (lane < 16) ? 0 : 16;
  const int koffB = (lane < 16) ? 0 : 16;
  const int rbase = (lane < 16) ? 0 : 8;

  for (int q = gwave; q < NQ; q += nwaves) {
    const int b = q / M_;
    const size_t rowBase = (size_t)b * N_;
    const float qx = new_xyz[q * 3 + 0];
    const float qy = new_xyz[q * 3 + 1];
    const float qz = new_xyz[q * 3 + 2];

    // ---- phase A: rel xyz + zero pad (k = 64..95), one lane per sample ----
    if (lane < S) {
      const int gi = idxArr[q * S + lane];
      float rx = 0.0f, ry = 0.0f, rz = 0.0f;
      if (gi >= 0) {
        const float* pp = xyz + (rowBase + (size_t)gi) * 3;
        rx = pp[0] - qx; ry = pp[1] - qy; rz = pp[2] - qz;
      }
      _Float16* row = Xa + lane * KP;
      row[64] = (_Float16)rx;
      row[65] = (_Float16)ry;
      row[66] = (_Float16)rz;
#pragma unroll
      for (int k = 67; k < KP; ++k) row[k] = (_Float16)0.0f;
    }

    // ---- phase B: gather features (k = 0..63), 8 lanes per sample ----
#pragma unroll
    for (int it = 0; it < S / 4; ++it) {
      const int s = it * 4 + (lane >> 3);
      const int j = (lane & 7) * 8;
      const int gi = idxArr[q * S + s];
      v8h hv;
      if (gi >= 0) {
        const float4* fp =
            (const float4*)(feats + (rowBase + (size_t)gi) * 64 + j);
        const float4 fa = fp[0];
        const float4 fb = fp[1];
        hv[0] = (_Float16)fa.x; hv[1] = (_Float16)fa.y;
        hv[2] = (_Float16)fa.z; hv[3] = (_Float16)fa.w;
        hv[4] = (_Float16)fb.x; hv[5] = (_Float16)fb.y;
        hv[6] = (_Float16)fb.z; hv[7] = (_Float16)fb.w;
      } else {
        hv = zero8h();
      }
      *(v8h*)(Xa + s * KP + j) = hv;
    }

    float cmax[4] = {-1e30f, -1e30f, -1e30f, -1e30f};

#pragma unroll
    for (int mt = 0; mt < S / 16; ++mt) {
      // -------- layer 1: (16 x 96) @ (96 x 64) --------
      v8f acc[4];
#pragma unroll
      for (int nt = 0; nt < 4; ++nt) acc[nt] = zero8f();

      const _Float16* arow = Xa + (mt * 16 + col) * KP;
#pragma unroll
      for (int ks = 0; ks < KP / 32; ++ks) {
        const int kb = ks * 32;
        const v8h alo = *(const v8h*)(arow + kb + koffA);
        const v8h ahi = *(const v8h*)(arow + kb + koffA + 8);
        const v16h a = cat16(alo, ahi);
#pragma unroll
        for (int nt = 0; nt < 4; ++nt) {
          const _Float16* brow = w1lds + (nt * 16 + col) * KP + kb + koffB;
          const v16h bb = cat16(*(const v8h*)brow, *(const v8h*)(brow + 8));
          acc[nt] = __builtin_amdgcn_wmma_f32_16x16x32_f16(
              false, a, false, bb, (short)0, acc[nt], false, false);
        }
      }

      // -------- BN + ReLU, stage layer-2 input tile (f16) --------
#pragma unroll
      for (int nt = 0; nt < 4; ++nt) {
        const int n = nt * 16 + col;
        const float sc = bns1[n];
        const float bi = bnb1[n];
#pragma unroll
        for (int r = 0; r < 8; ++r) {
          const float v = fmaxf(acc[nt][r] * sc + bi, 0.0f);
          X2[(rbase + r) * 64 + n] = (_Float16)v;
        }
      }

      // -------- layer 2: (16 x 64) @ (64 x 64) --------
      v8f acc2[4];
#pragma unroll
      for (int nt = 0; nt < 4; ++nt) acc2[nt] = zero8f();

      const _Float16* a2row = X2 + col * 64;
#pragma unroll
      for (int ks = 0; ks < 2; ++ks) {
        const int kb = ks * 32;
        const v8h alo = *(const v8h*)(a2row + kb + koffA);
        const v8h ahi = *(const v8h*)(a2row + kb + koffA + 8);
        const v16h a = cat16(alo, ahi);
#pragma unroll
        for (int nt = 0; nt < 4; ++nt) {
          const _Float16* brow = w2lds + (nt * 16 + col) * 64 + kb + koffB;
          const v16h bb = cat16(*(const v8h*)brow, *(const v8h*)(brow + 8));
          acc2[nt] = __builtin_amdgcn_wmma_f32_16x16x32_f16(
              false, a, false, bb, (short)0, acc2[nt], false, false);
        }
      }

      // -------- BN + ReLU + running max over this tile's 8 rows --------
#pragma unroll
      for (int nt = 0; nt < 4; ++nt) {
        const int n = nt * 16 + col;
        const float sc = bns2[n];
        const float bi = bnb2[n];
        float mx = cmax[nt];
#pragma unroll
        for (int r = 0; r < 8; ++r)
          mx = fmaxf(mx, fmaxf(acc2[nt][r] * sc + bi, 0.0f));
        cmax[nt] = mx;
      }
    }

    // ---- merge row-halves (lanes 0-15 hold M=0..7, 16-31 hold M=8..15) ----
#pragma unroll
    for (int nt = 0; nt < 4; ++nt) {
      const float other = __shfl_xor(cmax[nt], 16, 32);
      const float tot = fmaxf(cmax[nt], other);
      if (lane < 16)
        out_feat[(size_t)q * 128 + SCALE * 64 + nt * 16 + col] = tot;
    }
  }
}

// ---------------------------------------------------------------------
extern "C" void kernel_launch(void* const* d_in, const int* in_sizes, int n_in,
                              void* d_out, int out_size, void* d_ws,
                              size_t ws_size, hipStream_t stream) {
  (void)in_sizes; (void)n_in; (void)out_size; (void)ws_size;

  const float* xyz      = (const float*)d_in[0];
  const float* new_xyz  = (const float*)d_in[2];
  const float* feats    = (const float*)d_in[4];
  const float* W00 = (const float*)d_in[5];
  const float* g00 = (const float*)d_in[6];
  const float* b00 = (const float*)d_in[7];
  const float* W01 = (const float*)d_in[8];
  const float* g01 = (const float*)d_in[9];
  const float* b01 = (const float*)d_in[10];
  const float* W10 = (const float*)d_in[11];
  const float* g10 = (const float*)d_in[12];
  const float* b10 = (const float*)d_in[13];
  const float* W11 = (const float*)d_in[14];
  const float* g11 = (const float*)d_in[15];
  const float* b11 = (const float*)d_in[16];

  float* out      = (float*)d_out;
  float* out_xyz  = out;                       // [NQ*3]
  float* out_feat = out + (size_t)NQ * 3;      // [NQ*128]

  int* idx0 = (int*)d_ws;                      // [NQ*16]
  int* idx1 = idx0 + (size_t)NQ * NS0;         // [NQ*32]

  // ball query: one wave per query, 8 waves per 256-thread block
  ball_query_kernel<<<NQ / 8, 256, 0, stream>>>(xyz, new_xyz, idx0, idx1,
                                                out_xyz);

  constexpr int TPB = 128;                     // 4 waves/block
  constexpr int NW  = TPB / 32;
  constexpr size_t baseLds =
      (size_t)(64 * KP + 64 * 64) * 2 + 4 * 64 * 4;     // weights + BN
  constexpr size_t sm0 = baseLds + (size_t)NW * (NS0 * KP + 16 * 64) * 2;
  constexpr size_t sm1 = baseLds + (size_t)NW * (NS1 * KP + 16 * 64) * 2;

  sa_mlp_kernel<0><<<1024, TPB, sm0, stream>>>(xyz, new_xyz, feats, idx0,
                                               W00, g00, b00, W01, g01, b01,
                                               out_feat);
  sa_mlp_kernel<1><<<1024, TPB, sm1, stream>>>(xyz, new_xyz, feats, idx1,
                                               W10, g10, b10, W11, g11, b11,
                                               out_feat);
}
